// MSDAM_11948599017926
// MI455X (gfx1250) — matrix operand order repaired and checked by reading, not compiled, not run on hardware
//
#include <hip/hip_runtime.h>
#include <hip/hip_bf16.h>
#include <math.h>

typedef _Float16 v16h __attribute__((ext_vector_type(16)));
typedef _Float16 v8h  __attribute__((ext_vector_type(8)));
typedef float    v8f  __attribute__((ext_vector_type(8)));
typedef unsigned int uint32x4 __attribute__((ext_vector_type(4)));
typedef int int32x8 __attribute__((ext_vector_type(8)));
typedef int int32x4 __attribute__((ext_vector_type(4)));

// ---------------------------------------------------------------------------
// Tensor Data Mover: 1-D contiguous DMA of `nelem` f16 elements from global
// memory into LDS at byte offset `lds_off`. D# bitfields per
// CDNA5 ISA (group0: count=1, lds_addr, global_addr,
// type=2; group1: data_size=1(2B), tensor_dim0=tile_dim0=nelem, dim1=1).
// This toolchain exposes the 6-arg builtin (extra int32x8 group, then cpol).
// Completion via TENSORcnt.
// ---------------------------------------------------------------------------
__device__ __forceinline__ void tdm_load_1d_f16(const _Float16* gsrc,
                                                unsigned lds_off, unsigned nelem) {
  unsigned long long ga = (unsigned long long)(uintptr_t)gsrc;
  uint32x4 g0;
  g0[0] = 1u;                                         // count=1, user mode
  g0[1] = lds_off;                                    // lds_addr (bytes)
  g0[2] = (unsigned)(ga & 0xFFFFFFFFu);               // global_addr[31:0]
  g0[3] = (unsigned)((ga >> 32) & 0x1FFFFFFu) | (2u << 30);  // addr[56:32]|type=2
  int32x8 g1;
  g1[0] = (int)(1u << 16);                            // data_size=1 (2 bytes)
  g1[1] = (int)((nelem & 0xFFFFu) << 16);             // tensor_dim0 lo
  g1[2] = (int)(((nelem >> 16) & 0xFFFFu) | (1u << 16)); // dim0 hi | dim1=1
  g1[3] = (int)((nelem & 0xFFFFu) << 16);             // dim1 hi=0 | tile_dim0
  g1[4] = 1;                                          // tile_dim1=1, tile_dim2=0
  g1[5] = (int)nelem;                                 // dim0_stride lo
  g1[6] = (int)(((nelem >> 16) & 0xFFFFu) | ((nelem & 0xFFFFu) << 16));
  g1[7] = (int)((nelem >> 16) & 0xFFFFu);             // dim1_stride hi
  int32x4 z4 = {0, 0, 0, 0};                          // groups 2/3 unused (<=2D)
  int32x8 z8 = {0, 0, 0, 0, 0, 0, 0, 0};
  __builtin_amdgcn_tensor_load_to_lds(g0, g1, z4, z4, z8, 0);
}

// ---------------------------------------------------------------------------
// Weight pre-pack kernels: f32 -> f16 once per config, already transposed into
// the WMMA-B LDS layouts so the TDM copy is a plain linear move.
// ow (72,32,3,3) -> [t][o=80][c=32]  (o padded with zeros)
// ---------------------------------------------------------------------------
__global__ void pack_ow(const float* __restrict__ w, _Float16* __restrict__ out) {
  int i = blockIdx.x * 256 + threadIdx.x;           // 9*80*32 = 23040
  if (i >= 23040) return;
  int c = i & 31, o = (i >> 5) % 80, t = i / 2560;
  out[i] = (_Float16)((o < 72) ? w[((long long)o * 32 + c) * 9 + t] : 0.f);
}
// dw (64,8,3,3) -> [g][o=16][k=96], k = tap*8 + c (k padded with zeros)
__global__ void pack_dw(const float* __restrict__ w, _Float16* __restrict__ out) {
  int i = blockIdx.x * 256 + threadIdx.x;           // 4*16*96 = 6144
  if (i >= 6144) return;
  int k = i % 96, o = (i / 96) & 15, g = i / 1536;
  int kk = k >> 3, c = k & 7;
  out[i] = (_Float16)((kk < 9) ? w[(((long long)(g * 16 + o)) * 8 + c) * 9 + kk] : 0.f);
}

// ---------------------------------------------------------------------------
// avgpool (2,2)/2 with 1-px zero pad: [NC][64][64] -> [NC][33][33]
// ---------------------------------------------------------------------------
__global__ void avgpool2_kernel(const float* __restrict__ x, float* __restrict__ out,
                                int H, int W, int Hs, int Ws, long long total) {
  long long i = (long long)blockIdx.x * blockDim.x + threadIdx.x;
  if (i >= total) return;
  int wo = (int)(i % Ws);
  long long r = i / Ws;
  int ho = (int)(r % Hs);
  long long ch = r / Hs;
  const float* xc = x + ch * (long long)H * W;
  int y0 = 2 * ho - 1, x0 = 2 * wo - 1;
  float s = 0.f;
#pragma unroll
  for (int dy = 0; dy < 2; ++dy)
#pragma unroll
    for (int dx = 0; dx < 2; ++dx) {
      int yy = y0 + dy, xx = x0 + dx;
      if (yy >= 0 && yy < H && xx >= 0 && xx < W) s += xc[yy * W + xx];
    }
  out[i] = s * 0.25f;
}

// ---------------------------------------------------------------------------
// Offset conv: 3x3 dilated conv 32->72ch, implicit GEMM via WMMA.
// Block = 5 waves, one 16-row M tile shared by all 5 N tiles:
//   - weights [9][80][32] f16 staged to LDS by the Tensor Data Mover (wave 0)
//   - A tile [9][16][32] f16 staged cooperatively once per block
//   - per tap: A/B fragments = 2x ds_load_b128 each, one v_wmma_f32_16x16x32_f16
// ---------------------------------------------------------------------------
__global__ void __launch_bounds__(160) offset_conv_wmma(
    const float* __restrict__ X, const _Float16* __restrict__ Wh,
    const float* __restrict__ Bias, float* __restrict__ OFF,
    int Hs, int Ws, int dil, long long baseB, long long baseD, long long cstride) {
  const int HW = Hs * Ws;
  const int n  = blockIdx.z;
  const int m0 = blockIdx.x * 16;
  const int tid  = threadIdx.x;            // 0..159 (5 waves)
  const int wid  = tid >> 5;               // wave id == N tile
  const int lane = tid & 31;

  __shared__ _Float16 smem[9 * 80 * 32 + 9 * 16 * 32];   // Wlds | Alds
  _Float16* Wlds = smem;                   // [t][o(80)][c(32)]  (LDS offset 0)
  _Float16* Alds = smem + 9 * 80 * 32;     // [t][m(16)][c(32)]

  if (wid == 0) {                          // TDM weight stage + wait
    tdm_load_1d_f16(Wh, 0u, 9 * 80 * 32);
    __builtin_amdgcn_s_wait_tensorcnt(0);
  }

  const float* ximg = X + (long long)(n >> 3) * baseB + (long long)(n & 7) * baseD;
  for (int i = tid; i < 9 * 16 * 32; i += 160) {
    int c = i & 31, m = (i >> 5) & 15, t = i >> 9;
    int mg = m0 + m;
    int ypos = mg / Ws, xpos = mg % Ws;
    int yy = ypos + (t / 3 - 1) * dil, xx = xpos + (t % 3 - 1) * dil;
    bool ok = (mg < HW) && yy >= 0 && yy < Hs && xx >= 0 && xx < Ws;
    float v = ok ? ximg[(long long)c * cstride + (long long)yy * Ws + xx] : 0.f;
    Alds[t * 512 + m * 32 + c] = (_Float16)v;
  }
  __syncthreads();

  const int hf  = lane >> 4;
  const int col = lane & 15;
  const int row = lane & 15;
  const int o0  = wid * 16;
  const int oo  = o0 + col;

  v8f acc = {};
#pragma unroll
  for (int t = 0; t < 9; ++t) {
    const v8h* ap = (const v8h*)(Alds + t * 512 + row * 32);
    v8h alo = ap[2 * hf];                  // c = hf*16 .. hf*16+7
    v8h ahi = ap[2 * hf + 1];              // c = hf*16+8 .. hf*16+15
    const v8h* bp = (const v8h*)(Wlds + t * 2560 + oo * 32 + hf * 16);
    v8h blo = bp[0];                       // k = hf*16 .. +7
    v8h bhi = bp[1];                       // k = hf*16+8 .. +15
    v16h a, b;
#pragma unroll
    for (int j = 0; j < 8; ++j) {
      a[j] = alo[j]; a[8 + j] = ahi[j];
      b[j] = blo[j]; b[8 + j] = bhi[j];
    }
    acc = __builtin_amdgcn_wmma_f32_16x16x32_f16(false, a, false, b, (short)0, acc,
                                                 false, false);
  }

  if (oo < 72) {
    float* offimg = OFF + (long long)n * 72 * HW;
    float bias = Bias[oo];
#pragma unroll
    for (int v = 0; v < 8; ++v) {
      int m = m0 + v + 8 * hf;             // D: M = v + 8*laneHalf, N = col
      if (m < HW) offimg[(long long)oo * HW + m] = acc[v] + bias;
    }
  }
}

// ---------------------------------------------------------------------------
// Deformable conv (groups=4): bilinear gathers feed the WMMA A fragment;
// group weights [16][96] f16 staged via TDM; K=72 (padded 96) = 3 WMMAs.
// ---------------------------------------------------------------------------
__global__ void deform_conv_wmma(const float* __restrict__ X, const float* __restrict__ OFF,
                                 const _Float16* __restrict__ DWh, const float* __restrict__ DB,
                                 float* __restrict__ OUT, int Hs, int Ws, int dil,
                                 long long baseB, long long baseD, long long cstride) {
  const int HW = Hs * Ws;
  const int n  = blockIdx.z;
  const int g  = blockIdx.y;
  const int m0 = blockIdx.x * 16;
  const float* xg   = X + (long long)(n >> 3) * baseB + (long long)(n & 7) * baseD
                        + (long long)(g * 8) * cstride;
  const float* offp = OFF + ((long long)n * 72 + g * 18) * HW;
  float* outp = OUT + ((long long)n * 64 + g * 16) * HW;

  const int lane = threadIdx.x;            // 32 threads = 1 wave
  const int hf   = lane >> 4;
  const int col  = lane & 15;
  const int row  = lane & 15;

  __shared__ _Float16 Blds[16 * 96];       // [o][k], LDS offset 0
  tdm_load_1d_f16(DWh + (long long)g * 1536, 0u, 1536);
  __builtin_amdgcn_s_wait_tensorcnt(0);

  const int mg  = m0 + row;
  const bool mok = mg < HW;
  const int ypos = mok ? mg / Ws : 0;
  const int xpos = mok ? mg % Ws : 0;

  if (mok) {                               // prefetch the 18 offset planes
#pragma unroll
    for (int k = 0; k < 18; ++k)
      __builtin_prefetch(offp + (long long)k * HW + mg, 0, 1);
  }

  v8f acc = {};
#pragma unroll
  for (int s = 0; s < 3; ++s) {
    v16h a;
#pragma unroll
    for (int g16 = 0; g16 < 2; ++g16) {
      const int kk = s * 4 + hf * 2 + g16; // tap for this 8-element A group
      float vals[8];
#pragma unroll
      for (int c = 0; c < 8; ++c) vals[c] = 0.f;
      if (kk < 9 && mok) {
        float sy = offp[(long long)(kk * 2 + 0) * HW + mg] + (float)(ypos + (kk / 3 - 1) * dil);
        float sx = offp[(long long)(kk * 2 + 1) * HW + mg] + (float)(xpos + (kk % 3 - 1) * dil);
        float fy = floorf(sy), fx = floorf(sx);
        float ty = sy - fy, tx = sx - fx;
        int y0 = (int)fy, x0 = (int)fx, y1 = y0 + 1, x1 = x0 + 1;
        bool iy0 = (y0 >= 0) && (y0 < Hs), iy1 = (y1 >= 0) && (y1 < Hs);
        bool ix0 = (x0 >= 0) && (x0 < Ws), ix1 = (x1 >= 0) && (x1 < Ws);
        int y0c = min(max(y0, 0), Hs - 1), y1c = min(max(y1, 0), Hs - 1);
        int x0c = min(max(x0, 0), Ws - 1), x1c = min(max(x1, 0), Ws - 1);
        int i00 = y0c * Ws + x0c, i01 = y0c * Ws + x1c;
        int i10 = y1c * Ws + x0c, i11 = y1c * Ws + x1c;
        float w00 = (1.f - ty) * (1.f - tx), w01 = (1.f - ty) * tx;
        float w10 = ty * (1.f - tx),         w11 = ty * tx;
#pragma unroll
        for (int c = 0; c < 8; ++c) {
          const float* xc = xg + (long long)c * cstride;
          float s00 = (iy0 && ix0) ? xc[i00] : 0.f;
          float s01 = (iy0 && ix1) ? xc[i01] : 0.f;
          float s10 = (iy1 && ix0) ? xc[i10] : 0.f;
          float s11 = (iy1 && ix1) ? xc[i11] : 0.f;
          vals[c] = w00 * s00 + w01 * s01 + w10 * s10 + w11 * s11;
        }
      }
#pragma unroll
      for (int j = 0; j < 8; ++j) a[g16 * 8 + j] = (_Float16)vals[j];
    }
    const v8h* bp = (const v8h*)(Blds + col * 96 + s * 32 + hf * 16);
    v8h blo = bp[0], bhi = bp[1];
    v16h b;
#pragma unroll
    for (int j = 0; j < 8; ++j) { b[j] = blo[j]; b[8 + j] = bhi[j]; }
    acc = __builtin_amdgcn_wmma_f32_16x16x32_f16(false, a, false, b, (short)0, acc,
                                                 false, false);
  }

  float bias = DB[g * 16 + col];
#pragma unroll
  for (int v = 0; v < 8; ++v) {
    int m = m0 + v + 8 * hf;
    if (m < HW) outp[(long long)col * HW + m] = acc[v] + bias;
  }
}

// ---------------------------------------------------------------------------
// Instance norm (biased var, eps=1e-5) + exact GELU. One block per (img,ch).
// ---------------------------------------------------------------------------
__global__ void instnorm_gelu(const float* __restrict__ src, float* __restrict__ dst,
                              int HW) {
  const long long base = (long long)blockIdx.x * HW;
  const int tid = threadIdx.x;
  __shared__ float ssum[256], ssq[256];
  float s = 0.f, q = 0.f;
  for (int i = tid; i < HW; i += 256) {
    float v = src[base + i];
    s += v; q += v * v;
  }
  ssum[tid] = s; ssq[tid] = q;
  __syncthreads();
  for (int st = 128; st > 0; st >>= 1) {
    if (tid < st) { ssum[tid] += ssum[tid + st]; ssq[tid] += ssq[tid + st]; }
    __syncthreads();
  }
  float mean = ssum[0] / (float)HW;
  float var  = ssq[0] / (float)HW - mean * mean;
  float inv  = rsqrtf(var + 1e-5f);
  for (int i = tid; i < HW; i += 256) {
    float v = (src[base + i] - mean) * inv;
    dst[base + i] = 0.5f * v * (1.f + erff(v * 0.70710678118654752f));
  }
}

// ---------------------------------------------------------------------------
// Fused adaptive-pool(Hs,Ws -> 64,64) + cosine similarity over 512 channels.
// adapt-matrix rows computed arithmetically (identity when Hs==64).
// ---------------------------------------------------------------------------
__global__ void cos_sim(const float* __restrict__ bh, const float* __restrict__ bl,
                        float* __restrict__ sim, int Hs, int Ws, int cfg) {
  const int HW = Hs * Ws;
  const int pos = blockIdx.x;
  const int b = pos >> 12;
  const int yx = pos & 4095;
  const int y = yx >> 6, xo = yx & 63;
  const int ys = (y * Hs) >> 6,  ye = ((y + 1) * Hs + 63) >> 6;
  const int xs = (xo * Ws) >> 6, xe = ((xo + 1) * Ws + 63) >> 6;
  const float w = 1.f / (float)((ye - ys) * (xe - xs));

  const int tid = threadIdx.x;
  float dot = 0.f, nh = 0.f, nl = 0.f;
  for (int cd = tid; cd < 512; cd += 256) {
    int d = cd >> 6, c = cd & 63;
    long long off = ((long long)(b * 8 + d) * 64 + c) * HW;
    const float* ph = bh + off;
    const float* pl = bl + off;
    float hv = 0.f, lv = 0.f;
    for (int yi = ys; yi < ye; ++yi)
      for (int xi = xs; xi < xe; ++xi) {
        hv += ph[yi * Ws + xi];
        lv += pl[yi * Ws + xi];
      }
    hv *= w; lv *= w;
    dot += hv * lv; nh += hv * hv; nl += lv * lv;
  }
  __shared__ float sd[256], sh[256], sl[256];
  sd[tid] = dot; sh[tid] = nh; sl[tid] = nl;
  __syncthreads();
  for (int st = 128; st > 0; st >>= 1) {
    if (tid < st) { sd[tid] += sd[tid + st]; sh[tid] += sh[tid + st]; sl[tid] += sl[tid + st]; }
    __syncthreads();
  }
  if (tid == 0) {
    float nrm = sqrtf(sh[0]) * sqrtf(sl[0]);
    sim[((long long)b * 2 + cfg) * 4096 + yx] = sd[0] / fmaxf(nrm, 1e-8f);
  }
}

// ---------------------------------------------------------------------------
// Final fuse conv: (B,2,64,64) --3x3,pad1--> (B,128,64,64) + bias (K=18: VALU).
// ---------------------------------------------------------------------------
__global__ void final_conv(const float* __restrict__ cat, const float* __restrict__ fw,
                           const float* __restrict__ fb, float* __restrict__ out) {
  int i = blockIdx.x * 256 + threadIdx.x;
  if (i >= 2 * 128 * 4096) return;
  int xo = i & 63, y = (i >> 6) & 63, o = (i >> 12) & 127, b = i >> 19;
  float s = fb[o];
#pragma unroll
  for (int ci = 0; ci < 2; ++ci)
#pragma unroll
    for (int t = 0; t < 9; ++t) {
      int yy = y + t / 3 - 1, xx = xo + t % 3 - 1;
      if (yy >= 0 && yy < 64 && xx >= 0 && xx < 64)
        s += fw[(o * 2 + ci) * 9 + t] * cat[((long long)(b * 2 + ci) << 12) + yy * 64 + xx];
    }
  out[i] = s;
}

// ---------------------------------------------------------------------------
extern "C" void kernel_launch(void* const* d_in, const int* in_sizes, int n_in,
                              void* d_out, int out_size, void* d_ws, size_t ws_size,
                              hipStream_t stream) {
  (void)in_sizes; (void)n_in; (void)out_size; (void)ws_size;
  const float* hsi = (const float*)d_in[0];
  const float* lid = (const float*)d_in[1];
  const float* owc[2] = {(const float*)d_in[2], (const float*)d_in[6]};
  const float* obc[2] = {(const float*)d_in[3], (const float*)d_in[7]};
  const float* dwc[2] = {(const float*)d_in[4], (const float*)d_in[8]};
  const float* dbc[2] = {(const float*)d_in[5], (const float*)d_in[9]};
  const float* fw  = (const float*)d_in[10];
  const float* fb  = (const float*)d_in[11];
  float* out = (float*)d_out;

  // Workspace layout (floats/f16): ~52 MiB total
  float* ws     = (float*)d_ws;
  float* pooled = ws;                      // 16*32*1089      =   557,568 f32
  float* offb   = pooled + 557568;         // 16*72*4096      = 4,718,592 f32
  float* brH    = offb   + 4718592;        // 16*64*4096      = 4,194,304 f32
  float* brL    = brH    + 4194304;        // 16*64*4096      = 4,194,304 f32
  float* simb   = brL    + 4194304;        // 2*2*4096        =    16,384 f32
  _Float16* whP = (_Float16*)(simb + 16384);   // 23,040 f16 (packed offset wts)
  _Float16* dwP = whP + 23040;                 //  6,144 f16 (packed deform wts)

  const float* srcs[2] = {hsi, lid};
  float* brs[2] = {brH, brL};

  for (int cfg = 0; cfg < 2; ++cfg) {
    const int Hs = cfg ? 64 : 33, Ws = Hs, HW = Hs * Ws;
    const int dil = cfg ? 2 : 1;
    const long long baseB = 32LL * 8 * HW, baseD = HW, cstride = 8LL * HW;
    const int mtiles = (HW + 15) / 16;

    pack_ow<<<(23040 + 255) / 256, 256, 0, stream>>>(owc[cfg], whP);
    pack_dw<<<(6144 + 255) / 256, 256, 0, stream>>>(dwc[cfg], dwP);

    for (int m = 0; m < 2; ++m) {
      const float* src = srcs[m];
      if (cfg == 0) {
        avgpool2_kernel<<<(557568 + 255) / 256, 256, 0, stream>>>(
            srcs[m], pooled, 64, 64, Hs, Ws, 557568LL);
        src = pooled;
      }
      offset_conv_wmma<<<dim3(mtiles, 1, 16), 160, 0, stream>>>(
          src, whP, obc[cfg], offb, Hs, Ws, dil, baseB, baseD, cstride);
      deform_conv_wmma<<<dim3(mtiles, 4, 16), 32, 0, stream>>>(
          src, offb, dwP, dbc[cfg], brs[m], Hs, Ws, dil, baseB, baseD, cstride);
      instnorm_gelu<<<16 * 64, 256, 0, stream>>>(brs[m], brs[m], HW);
    }
    cos_sim<<<2 * 4096, 256, 0, stream>>>(brH, brL, simb, Hs, Ws, cfg);
  }

  final_conv<<<(2 * 128 * 4096 + 255) / 256, 256, 0, stream>>>(simb, fw, fb, out);
  instnorm_gelu<<<2 * 128, 256, 0, stream>>>(out, out, 4096);
}
